// InvertedResidual_b_54125177864370
// MI455X (gfx1250) — 8-bit matrix operand order repaired and checked by reading, compiled, not run on hardware
//
#include <hip/hip_runtime.h>
#include <stdint.h>

typedef int v8i __attribute__((ext_vector_type(8)));

#define P_TOT   100352L
#define HW      3136
#define WDIM    56
#define EPSF    1e-5f
#define STAT_PB 2048L

__device__ __forceinline__ int koffA(int v, int half) {
  return ((v & 3) << 2) | (half << 4) | ((v >> 2) << 5);
}
__device__ __forceinline__ int sgnf(float v) { return (v > 0.f) - (v < 0.f); }
__device__ __forceinline__ uint32_t pk8(int a, int b, int c, int d) {
  return (uint32_t)(uint8_t)(int8_t)a | ((uint32_t)(uint8_t)(int8_t)b << 8) |
         ((uint32_t)(uint8_t)(int8_t)c << 16) | ((uint32_t)(uint8_t)(int8_t)d << 24);
}
__device__ __forceinline__ uint32_t pk16(int a, int b) {
  return (uint32_t)(uint16_t)(int16_t)a | ((uint32_t)(uint16_t)(int16_t)b << 16);
}
__device__ __forceinline__ v8i loadB(const uint32_t* bp, int kb, int half) {
  int di0 = kb * 16 + half * 4;
  v8i B;
  B[0] = bp[di0 + 0]; B[1] = bp[di0 + 1]; B[2] = bp[di0 + 2];  B[3] = bp[di0 + 3];
  B[4] = bp[di0 + 8]; B[5] = bp[di0 + 9]; B[6] = bp[di0 + 10]; B[7] = bp[di0 + 11];
  return B;
}

__global__ void k_zero(unsigned long long* p, int n) {
  int t = blockIdx.x * blockDim.x + threadIdx.x;
  if (t < n) p[t] = 0ULL;
}

__global__ void k_sf(const float* __restrict__ w, float* __restrict__ sf, int O, int K) {
  int o = blockIdx.x * blockDim.x + threadIdx.x;
  if (o >= O) return;
  float s = 0.f;
  for (int k = 0; k < K; ++k) s += fabsf(w[(long)o * K + k]);
  sf[o] = s / (float)K;
}

__global__ void k_prep_a1(const float* __restrict__ w1, uint32_t* __restrict__ ap) {
  int t = blockIdx.x * blockDim.x + threadIdx.x;
  if (t >= 16 * 32) return;
  int tile = t >> 5, lane = t & 31;
  int m = lane & 15, half = lane >> 4;
  int o = tile * 16 + m;
  for (int v = 0; v < 8; ++v) {
    int kb0 = koffA(v, half);
    uint32_t d = 0;
    for (int b = 0; b < 4; ++b)
      d |= (uint32_t)(uint8_t)(int8_t)sgnf(w1[o * 64 + kb0 + b]) << (8 * b);
    ap[t * 8 + v] = d;
  }
}

__global__ void k_prep_a2(const float* __restrict__ w2, uint32_t* __restrict__ ap) {
  int t = blockIdx.x * blockDim.x + threadIdx.x;
  if (t >= 576 * 32) return;
  int tile = t >> 5, lane = t & 31;
  int mt = tile & 15, kb = (tile >> 4) & 3, tap = tile >> 6;
  int m = lane & 15, half = lane >> 4;
  int o = mt * 16 + m;
  for (int v = 0; v < 8; ++v) {
    int kb0 = koffA(v, half);
    uint32_t d = 0;
    for (int b = 0; b < 4; ++b) {
      int i = kb * 64 + kb0 + b;
      d |= (uint32_t)(uint8_t)(int8_t)sgnf(w2[(long)(o * 256 + i) * 9 + tap]) << (8 * b);
    }
    ap[t * 8 + v] = d;
  }
}

__global__ void k_prep_a3(const float* __restrict__ w3, uint32_t* __restrict__ ap) {
  int t = blockIdx.x * blockDim.x + threadIdx.x;
  if (t >= 16 * 32) return;
  int tile = t >> 5, lane = t & 31;
  int mt = tile & 3, kb = tile >> 2;
  int m = lane & 15, half = lane >> 4;
  int o = mt * 16 + m;
  for (int v = 0; v < 8; ++v) {
    int kb0 = koffA(v, half);
    uint32_t d = 0;
    for (int b = 0; b < 4; ++b)
      d |= (uint32_t)(uint8_t)(int8_t)sgnf(w3[o * 256 + kb * 64 + kb0 + b]) << (8 * b);
    ap[t * 8 + v] = d;
  }
}

__global__ void k_binx(const float* __restrict__ x, int8_t* __restrict__ xb) {
  long t = (long)blockIdx.x * blockDim.x + threadIdx.x;
  if (t >= 32L * 64 * HW) return;
  long s = t % HW;
  long nc = t / HW;
  int c = (int)(nc & 63);
  int n = (int)(nc >> 6);
  xb[((long)n * HW + s) * 64 + c] = (int8_t)sgnf(x[t]);
}

__global__ __launch_bounds__(128) void k_conv1(const int8_t* __restrict__ xb,
                                               const int* __restrict__ ap1,
                                               int8_t* __restrict__ y1b) {
  int lane = threadIdx.x & 31;
  int tile = blockIdx.x * (blockDim.x >> 5) + (threadIdx.x >> 5);
  int col = lane & 15, half = lane >> 4;
  long pix = (long)tile * 16 + col;
  const uint32_t* bp = (const uint32_t*)(xb + pix * 64);
  v8i B = loadB(bp, 0, half);
  for (int g = 0; g < 4; ++g) {
    v8i A[4], acc[4];
#pragma unroll
    for (int j = 0; j < 4; ++j)
      A[j] = *(const v8i*)(ap1 + ((g * 4 + j) * 32 + lane) * 8);
#pragma unroll
    for (int j = 0; j < 4; ++j) {
      v8i z = {0, 0, 0, 0, 0, 0, 0, 0};
      acc[j] = __builtin_amdgcn_wmma_i32_16x16x64_iu8(true, A[j], true, B, z, false, false);
    }
#pragma unroll
    for (int j = 0; j < 4; ++j) {
      int mt = g * 4 + j;
      uint2 st;
      st.x = pk8(acc[j][0], acc[j][1], acc[j][2], acc[j][3]);
      st.y = pk8(acc[j][4], acc[j][5], acc[j][6], acc[j][7]);
      *(uint2*)(y1b + pix * 256 + mt * 16 + half * 8) = st;
    }
  }
}

__global__ __launch_bounds__(128) void k_conv2(const int8_t* __restrict__ a2,
                                               const int* __restrict__ ap2,
                                               int16_t* __restrict__ y2h) {
  int lane = threadIdx.x & 31;
  int job = blockIdx.x * (blockDim.x >> 5) + (threadIdx.x >> 5);
  int tile = job >> 1, mhalf = job & 1;
  int col = lane & 15, half = lane >> 4;
  long p = (long)tile * 16 + col;
  int n = (int)(p / HW);
  int rem = (int)(p % HW);
  int h = rem / WDIM, w = rem % WDIM;
  v8i acc[8];
#pragma unroll
  for (int i = 0; i < 8; ++i) acc[i] = (v8i){0, 0, 0, 0, 0, 0, 0, 0};
  for (int tap = 0; tap < 9; ++tap) {
    int hh = h + tap / 3 - 1, ww = w + tap % 3 - 1;
    bool valid = ((unsigned)hh < (unsigned)WDIM) && ((unsigned)ww < (unsigned)WDIM);
    const uint32_t* bp = (const uint32_t*)(a2 + ((long)n * HW + hh * WDIM + ww) * 256);
#pragma unroll
    for (int kb = 0; kb < 4; ++kb) {
      v8i B = {0, 0, 0, 0, 0, 0, 0, 0};
      if (valid) B = loadB(bp, kb, half);
      v8i A[8];
#pragma unroll
      for (int mi = 0; mi < 8; ++mi) {
        int tileA = (tap * 4 + kb) * 16 + (mhalf * 8 + mi);
        A[mi] = *(const v8i*)(ap2 + (tileA * 32 + lane) * 8);
      }
#pragma unroll
      for (int mi = 0; mi < 8; ++mi)
        acc[mi] = __builtin_amdgcn_wmma_i32_16x16x64_iu8(true, A[mi], true, B, acc[mi], false, false);
    }
  }
#pragma unroll
  for (int mi = 0; mi < 8; ++mi) {
    int mt = mhalf * 8 + mi;
    uint4 st;
    st.x = pk16(acc[mi][0], acc[mi][1]);
    st.y = pk16(acc[mi][2], acc[mi][3]);
    st.z = pk16(acc[mi][4], acc[mi][5]);
    st.w = pk16(acc[mi][6], acc[mi][7]);
    *(uint4*)(y2h + p * 256 + mt * 16 + half * 8) = st;
  }
}

__global__ __launch_bounds__(128) void k_conv3(const int8_t* __restrict__ a3,
                                               const int* __restrict__ ap3,
                                               int16_t* __restrict__ y3h) {
  int lane = threadIdx.x & 31;
  int tile = blockIdx.x * (blockDim.x >> 5) + (threadIdx.x >> 5);
  int col = lane & 15, half = lane >> 4;
  long p = (long)tile * 16 + col;
  const uint32_t* bp = (const uint32_t*)(a3 + p * 256);
  v8i acc[4];
#pragma unroll
  for (int i = 0; i < 4; ++i) acc[i] = (v8i){0, 0, 0, 0, 0, 0, 0, 0};
#pragma unroll
  for (int kb = 0; kb < 4; ++kb) {
    v8i B = loadB(bp, kb, half);
    v8i A[4];
#pragma unroll
    for (int mt = 0; mt < 4; ++mt)
      A[mt] = *(const v8i*)(ap3 + ((kb * 4 + mt) * 32 + lane) * 8);
#pragma unroll
    for (int mt = 0; mt < 4; ++mt)
      acc[mt] = __builtin_amdgcn_wmma_i32_16x16x64_iu8(true, A[mt], true, B, acc[mt], false, false);
  }
#pragma unroll
  for (int mt = 0; mt < 4; ++mt) {
    uint4 st;
    st.x = pk16(acc[mt][0], acc[mt][1]);
    st.y = pk16(acc[mt][2], acc[mt][3]);
    st.z = pk16(acc[mt][4], acc[mt][5]);
    st.w = pk16(acc[mt][6], acc[mt][7]);
    *(uint4*)(y3h + p * 64 + mt * 16 + half * 8) = st;
  }
}

__global__ void k_stats8(const int8_t* __restrict__ y, int C, long P,
                         unsigned long long* __restrict__ st) {
  int c = threadIdx.x;
  long p0 = (long)blockIdx.x * STAT_PB;
  long p1 = p0 + STAT_PB; if (p1 > P) p1 = P;
  long long s = 0, sq = 0;
  for (long p = p0; p < p1; ++p) { int v = y[p * C + c]; s += v; sq += v * v; }
  atomicAdd(&st[c], (unsigned long long)s);
  atomicAdd(&st[C + c], (unsigned long long)sq);
}

__global__ void k_stats16(const int16_t* __restrict__ y, int C, long P,
                          unsigned long long* __restrict__ st) {
  int c = threadIdx.x;
  long p0 = (long)blockIdx.x * STAT_PB;
  long p1 = p0 + STAT_PB; if (p1 > P) p1 = P;
  long long s = 0, sq = 0;
  for (long p = p0; p < p1; ++p) { int v = y[p * C + c]; s += v; sq += (long long)v * v; }
  atomicAdd(&st[c], (unsigned long long)s);
  atomicAdd(&st[C + c], (unsigned long long)sq);
}

__device__ __forceinline__ void bn_params(const unsigned long long* st, int C, int c,
                                          float s, float& mu, float& rstd) {
  float invP = 1.f / (float)P_TOT;
  float ma = (float)(long long)st[c] * invP;
  float ea2 = (float)(long long)st[C + c] * invP;
  mu = s * ma;
  float var = s * s * (ea2 - ma * ma);
  rstd = rsqrtf(var + EPSF);
}

__global__ void k_bnsign1(const int8_t* __restrict__ y1b, const unsigned long long* __restrict__ st,
                          const float* __restrict__ sf1, const float* __restrict__ g1,
                          const float* __restrict__ b1, int8_t* __restrict__ a2) {
  long t = (long)blockIdx.x * blockDim.x + threadIdx.x;
  if (t >= P_TOT * 256) return;
  int c = (int)(t & 255);
  float s = sf1[c], mu, rstd;
  bn_params(st, 256, c, s, mu, rstd);
  float v = ((float)y1b[t] * s - mu) * rstd * g1[c] + b1[c];
  a2[t] = (int8_t)sgnf(v);
}

__global__ void k_sepool(const int16_t* __restrict__ y2h, const unsigned long long* __restrict__ st,
                         const float* __restrict__ sf2, const float* __restrict__ g2,
                         const float* __restrict__ b2, float* __restrict__ pooled) {
  int n = blockIdx.x, c = threadIdx.x;
  float s = sf2[c], mu, rstd;
  bn_params(st, 256, c, s, mu, rstd);
  float g = g2[c], b = b2[c], acc = 0.f;
  long base = (long)n * HW;
  for (int i = 0; i < HW; ++i)
    acc += ((float)y2h[(base + i) * 256 + c] * s - mu) * rstd * g + b;
  pooled[n * 256 + c] = acc * (1.f / (float)HW);
}

__global__ void k_semlp(const float* __restrict__ pooled,
                        const float* __restrict__ w1, const float* __restrict__ b1,
                        const float* __restrict__ w2, const float* __restrict__ b2,
                        float* __restrict__ yscale) {
  __shared__ float hid[64];
  __shared__ float pv[256];
  int n = blockIdx.x, t = threadIdx.x;
  pv[t] = pooled[n * 256 + t];
  __syncthreads();
  if (t < 64) {
    float s = b1[t];
    for (int k = 0; k < 256; ++k) s += w1[t * 256 + k] * pv[k];
    hid[t] = fmaxf(s, 0.f);
  }
  __syncthreads();
  float s = b2[t];
  for (int j = 0; j < 64; ++j) s += w2[t * 64 + j] * hid[j];
  yscale[n * 256 + t] = fminf(fmaxf(s + 3.f, 0.f), 6.f) * (1.f / 6.f);
}

__global__ void k_bnsign2(const int16_t* __restrict__ y2h, const unsigned long long* __restrict__ st,
                          const float* __restrict__ sf2, const float* __restrict__ g2,
                          const float* __restrict__ b2, const float* __restrict__ yscale,
                          int8_t* __restrict__ a3) {
  long t = (long)blockIdx.x * blockDim.x + threadIdx.x;
  if (t >= P_TOT * 256) return;
  int c = (int)(t & 255);
  long p = t >> 8;
  int n = (int)(p / HW);
  float s = sf2[c], mu, rstd;
  bn_params(st, 256, c, s, mu, rstd);
  float v = ((float)y2h[t] * s - mu) * rstd * g2[c] + b2[c];
  v *= yscale[n * 256 + c];
  a3[t] = (int8_t)sgnf(v);
}

__global__ void k_final(const float* __restrict__ x, const int16_t* __restrict__ y3h,
                        const unsigned long long* __restrict__ st,
                        const float* __restrict__ sf3, const float* __restrict__ g3,
                        const float* __restrict__ b3, float* __restrict__ out) {
  long t = (long)blockIdx.x * blockDim.x + threadIdx.x;
  if (t >= 32L * 64 * HW) return;
  long s = t % HW;
  long nc = t / HW;
  int c = (int)(nc & 63);
  int n = (int)(nc >> 6);
  float sf = sf3[c], mu, rstd;
  bn_params(st, 64, c, sf, mu, rstd);
  int acc = y3h[((long)n * HW + s) * 64 + c];
  float v = ((float)acc * sf - mu) * rstd * g3[c] + b3[c];
  out[t] = x[t] + v;
}

extern "C" void kernel_launch(void* const* d_in, const int* in_sizes, int n_in,
                              void* d_out, int out_size, void* d_ws, size_t ws_size,
                              hipStream_t stream) {
  const float* x    = (const float*)d_in[0];
  const float* w1   = (const float*)d_in[1];
  const float* g1   = (const float*)d_in[2];
  const float* b1   = (const float*)d_in[3];
  const float* w2   = (const float*)d_in[4];
  const float* g2   = (const float*)d_in[5];
  const float* b2   = (const float*)d_in[6];
  const float* sew1 = (const float*)d_in[7];
  const float* seb1 = (const float*)d_in[8];
  const float* sew2 = (const float*)d_in[9];
  const float* seb2 = (const float*)d_in[10];
  const float* w3   = (const float*)d_in[11];
  const float* g3   = (const float*)d_in[12];
  const float* b3   = (const float*)d_in[13];

  char* ws = (char*)d_ws;
  size_t off = 0;
  auto take = [&](size_t bytes) -> void* {
    void* p = ws + off;
    off += (bytes + 255) & ~(size_t)255;
    return p;
  };
  int8_t*  xb  = (int8_t*)take(P_TOT * 64);
  int8_t*  y1b = (int8_t*)take(P_TOT * 256);
  int8_t*  a2  = (int8_t*)take(P_TOT * 256);
  int16_t* y2h = (int16_t*)take(P_TOT * 256 * 2);
  int16_t* y3h = (int16_t*)take(P_TOT * 64 * 2);
  int* ap1 = (int*)take(16 * 1024);
  int* ap2 = (int*)take(576 * 1024);
  int* ap3 = (int*)take(16 * 1024);
  float* sf1 = (float*)take(256 * 4);
  float* sf2 = (float*)take(256 * 4);
  float* sf3 = (float*)take(64 * 4);
  unsigned long long* stats = (unsigned long long*)take(1152 * 8);
  float* pooled = (float*)take(32 * 256 * 4);
  float* yscale = (float*)take(32 * 256 * 4);
  int8_t* a3 = y1b;
  unsigned long long* st1 = stats;
  unsigned long long* st2 = stats + 512;
  unsigned long long* st3 = stats + 1024;

  const long NTOT = 32L * 64 * HW;
  const int statBlocks = (int)((P_TOT + STAT_PB - 1) / STAT_PB);

  k_zero<<<(1152 + 255) / 256, 256, 0, stream>>>(stats, 1152);
  k_sf<<<1, 256, 0, stream>>>(w1, sf1, 256, 64);
  k_sf<<<1, 256, 0, stream>>>(w2, sf2, 256, 2304);
  k_sf<<<1, 64, 0, stream>>>(w3, sf3, 64, 256);
  k_prep_a1<<<2, 256, 0, stream>>>(w1, (uint32_t*)ap1);
  k_prep_a2<<<72, 256, 0, stream>>>(w2, (uint32_t*)ap2);
  k_prep_a3<<<2, 256, 0, stream>>>(w3, (uint32_t*)ap3);
  k_binx<<<(int)((NTOT + 255) / 256), 256, 0, stream>>>(x, xb);

  k_conv1<<<6272 / 4, 128, 0, stream>>>(xb, ap1, y1b);
  k_stats8<<<statBlocks, 256, 0, stream>>>(y1b, 256, P_TOT, st1);
  k_bnsign1<<<(int)((P_TOT * 256 + 255) / 256), 256, 0, stream>>>(y1b, st1, sf1, g1, b1, a2);

  k_conv2<<<(6272 * 2) / 4, 128, 0, stream>>>(a2, ap2, y2h);
  k_stats16<<<statBlocks, 256, 0, stream>>>(y2h, 256, P_TOT, st2);
  k_sepool<<<32, 256, 0, stream>>>(y2h, st2, sf2, g2, b2, pooled);
  k_semlp<<<32, 256, 0, stream>>>(pooled, sew1, seb1, sew2, seb2, yscale);
  k_bnsign2<<<(int)((P_TOT * 256 + 255) / 256), 256, 0, stream>>>(y2h, st2, sf2, g2, b2, yscale, a3);

  k_conv3<<<6272 / 4, 128, 0, stream>>>(a3, ap3, y3h);
  k_stats16<<<statBlocks, 64, 0, stream>>>(y3h, 64, P_TOT, st3);
  k_final<<<(int)((NTOT + 255) / 256), 256, 0, stream>>>(x, y3h, st3, sf3, g3, b3, (float*)d_out);
  (void)in_sizes; (void)n_in; (void)out_size; (void)ws_size;
}
